// G_synthesis_stylegan2_36584531427336
// MI455X (gfx1250) — matrix operand order repaired and checked by reading, not compiled, not run on hardware
//
#include <hip/hip_runtime.h>
#include <math.h>

// ---------------------------------------------------------------------------
// StyleGAN2-like synthesis network for MI455X (gfx1250, wave32, WMMA).
//
//  * Conv contractions: implicit GEMM on v_wmma_f32_16x16x32_f16, one wave
//    per 16(out-ch) x 16(pixel) tile, K = in-ch x taps in chunks of 32,
//    fp32 accumulation. Kernel templated on <KS, DIL> so the tap loop fully
//    unrolls (9 WMMAs per K-chunk for 3x3).
//  * Weight tiles stream global->LDS with GLOBAL_LOAD_ASYNC_TO_LDS_B128
//    (ASYNCcnt), double-buffered: next K-chunk's weights are in flight while
//    the current chunk's WMMAs execute; s_wait_asynccnt gates buffer reuse.
//  * Activations NHWC f16, channels padded to a multiple of 32 -> every WMMA
//    A/B fragment is two contiguous 16B loads (A and B share contiguous k = 16*hi + s order;
//    f32 D in ISA 16x16 VGPR layout). LDS rows padded to 80B: aligned
//    b128 reads, conflict-free bank mapping.
//  * Style modulation (f16) folded into the B fragment; demod, noise, bias,
//    lrelu*sqrt2 fused into epilogues. Zero-padded weight rows/cols keep
//    padded channels inert -> no buffer clearing between graph replays.
// ---------------------------------------------------------------------------

typedef __attribute__((ext_vector_type(16))) _Float16 v16h;
typedef __attribute__((ext_vector_type(8)))  float    v8f;

#define THREADS 256

static inline int idiv(long a, long b) { return (int)((a + b - 1) / b); }

__device__ __forceinline__ float ftap(int i) {
  return (i == 0 || i == 3) ? 0.125f : 0.375f;   // [1,3,3,1]/8
}
__device__ __forceinline__ float lrelu_s2(float v) {
  return (v < 0.f ? 0.2f * v : v) * 1.41421356237309515f;
}

// ---- f32 NCHW -> f16 NHWC channel segment (zero pads channels >= C) -------
__global__ void k_pack(const float* __restrict__ src, _Float16* __restrict__ dst,
                       int B, int C, int H, int W, int Cpad, int c_off, int Cseg,
                       int bcast) {
  long i = (long)blockIdx.x * blockDim.x + threadIdx.x;
  long total = (long)B * H * W * Cseg;
  if (i >= total) return;
  int c = (int)(i % Cseg); long p = i / Cseg;
  int x = (int)(p % W); p /= W;
  int y = (int)(p % H); int b = (int)(p / H);
  float v = 0.f;
  if (c < C) {
    int bs = bcast ? 0 : b;
    v = src[(((long)bs * C + c) * H + y) * W + x];
  }
  dst[(((long)b * H + y) * W + x) * Cpad + c_off + c] = (_Float16)v;
}

// ---- f16 NHWC -> f16 NHWC channel-range copy (for concat) -----------------
__global__ void k_copy_h(const _Float16* __restrict__ src, _Float16* __restrict__ dst,
                         int B, int H, int W, int Cs_pad, int Cd_pad, int c_off,
                         int Cseg) {
  long i = (long)blockIdx.x * blockDim.x + threadIdx.x;
  long total = (long)B * H * W * Cseg;
  if (i >= total) return;
  int c = (int)(i % Cseg); long p = i / Cseg;
  int x = (int)(p % W); p /= W;
  int y = (int)(p % H); int b = (int)(p / H);
  dst[(((long)b * H + y) * W + x) * Cd_pad + c_off + c] =
      src[(((long)b * H + y) * W + x) * Cs_pad + c];
}

// ---- depthwise 4x4 FIR, NHWC f16, optional fused noise/bias/act -----------
__global__ void k_fir_h(const _Float16* __restrict__ X, _Float16* __restrict__ Y,
                        int B, int Hin, int Win, int Cpad, int Hout, int Wout,
                        int pad, float gain,
                        const float* __restrict__ noise, const float* __restrict__ nstr,
                        const float* __restrict__ bias, int Creal, int act) {
  long i = (long)blockIdx.x * blockDim.x + threadIdx.x;
  long total = (long)B * Hout * Wout * Cpad;
  if (i >= total) return;
  int c = (int)(i % Cpad); long p = i / Cpad;
  int x = (int)(p % Wout); p /= Wout;
  int y = (int)(p % Hout); int b = (int)(p / Hout);
  float acc = 0.f;
#pragma unroll
  for (int ty = 0; ty < 4; ++ty) {
    int iy = y - pad + ty;
    if (iy < 0 || iy >= Hin) continue;
    float fy = ftap(ty);
#pragma unroll
    for (int tx = 0; tx < 4; ++tx) {
      int ix = x - pad + tx;
      if (ix < 0 || ix >= Win) continue;
      acc += fy * ftap(tx) * (float)X[(((long)b * Hin + iy) * Win + ix) * Cpad + c];
    }
  }
  acc *= gain;
  if (noise) acc += noise[y * Wout + x] * nstr[0];
  if (bias && c < Creal) acc += bias[c];
  if (act) acc = lrelu_s2(acc);
  Y[(((long)b * Hout + y) * Wout + x) * Cpad + c] = (_Float16)acc;
}

// ---- implicit-GEMM conv on WMMA f16->f32, async-LDS weight pipeline -------
// X : NHWC f16 [B,Hin,Win,Ipad], Wt : f16 [KS*KS][Opad][Ipad]
// Yh: NHWC f16 out (or null), Yf: NCHW f32 out (or null)
// style16 [B,Ipad] f16 multiplies the B fragment (modulation), demod [B,Opad]
// scales the output; noise/bias/lrelu fused. DIL==2 => lhs dilation (up-conv).
template <int KS, int DIL>
__global__ void __launch_bounds__(256)
k_conv_wmma(const _Float16* __restrict__ X, const _Float16* __restrict__ Wt,
            _Float16* __restrict__ Yh, float* __restrict__ Yf,
            const _Float16* __restrict__ style16, const float* __restrict__ demod,
            const float* __restrict__ bias,
            const float* __restrict__ noise, const float* __restrict__ nstr,
            int B, int Hin, int Win, int Ipad,
            int Hout, int Wout, int Opad, int Oreal,
            int stride, int pad, int act) {
  (void)B;
  constexpr int TAPS = KS * KS;
  constexpr int LROW = 40;                       // halves; 80B row stride
  constexpr int BUFE = TAPS * 16 * LROW;         // halves per LDS buffer
  constexpr int SEGS = TAPS * 16 * 4;            // 16B segments per stage
  constexpr int STAGE_IT = (SEGS + 255) / 256;   // async instrs per thread
  __shared__ _Float16 wlds[2 * BUFE];            // double buffer

  const int wid  = threadIdx.x >> 5;
  const int lane = threadIdx.x & 31;
  const int HW = Hout * Wout;
  const int ntiles = (HW + 15) >> 4;
  int tile = blockIdx.x * 8 + wid;
  if (tile >= ntiles) tile = ntiles - 1;         // clamp (dup writes benign)
  const int b  = blockIdx.z;
  const int ot = blockIdx.y;                     // 16-wide out-channel tile
  const int nlo = lane & 15;
  const int hi  = lane >> 4;

  const int n  = tile * 16 + nlo;
  const int nc = (n < HW) ? n : (HW - 1);
  const int oy = nc / Wout, ox = nc - (nc / Wout) * Wout;

  // async DMA of one K-chunk of the weight tile into LDS buffer `buf`.
  // Uniform trip count (clamped seg -> duplicate load) so every wave issues
  // exactly STAGE_IT async instructions -> deterministic s_wait_asynccnt.
  auto stage = [&](int buf, int ic) {
#pragma unroll
    for (int it = 0; it < STAGE_IT; ++it) {
      int s = (int)threadIdx.x + it * 256;
      if (s >= SEGS) s = SEGS - 1;
      const int seg = s & 3;                     // 16B segment within row
      const int row = (s >> 2) & 15;
      const int t   = s >> 6;
      const unsigned ldsoff =
          (unsigned)((buf * BUFE + (t * 16 + row) * LROW + 8 * seg) * 2);
      const unsigned long long ga = (unsigned long long)(size_t)(
          Wt + ((long)t * Opad + ot * 16 + row) * Ipad + ic + 8 * seg);
      asm volatile("global_load_async_to_lds_b128 %0, %1, off"
                   :: "v"(ldsoff), "v"(ga) : "memory");
    }
  };

  v8f acc = {0.f, 0.f, 0.f, 0.f, 0.f, 0.f, 0.f, 0.f};

  const int nchunks = Ipad >> 5;
  stage(0, 0);                                   // prologue: chunk 0 in flight
  for (int ci = 0; ci < nchunks; ++ci) {
    const int ic  = ci << 5;
    const int cur = ci & 1;
    const bool more = (ci + 1 < nchunks);

    __syncthreads();                             // readers of other buf done
    if (more) stage(cur ^ 1, ic + 32);           // prefetch next chunk
    if (more) {                                  // wait: cur landed, next flies
      if constexpr (STAGE_IT == 3) asm volatile("s_wait_asynccnt 3" ::: "memory");
      else                         asm volatile("s_wait_asynccnt 1" ::: "memory");
    } else {
      asm volatile("s_wait_asynccnt 0" ::: "memory");
    }
    __syncthreads();                             // all waves' cur data in LDS

    // per-lane style slice (B-fragment channels ic + 16*hi ..)
    v16h sf;
    if (style16) {
      const _Float16* sp = style16 + (long)b * Ipad + ic + 16 * hi;
      ((uint4*)&sf)[0] = *(const uint4*)(sp);
      ((uint4*)&sf)[1] = *(const uint4*)(sp + 8);
    }

#pragma unroll
    for (int t = 0; t < TAPS; ++t) {
      const int ty = t / KS, tx = t % KS;        // compile-time

      // A fragment from LDS (contiguous k = 16*hi + s order, same as B).
      v16h a;
      const _Float16* wl = wlds + cur * BUFE + (t * 16 + nlo) * LROW;
      ((uint4*)&a)[0] = *(const uint4*)(wl + 16 * hi);      // K = 16hi..16hi+7
      ((uint4*)&a)[1] = *(const uint4*)(wl + 16 * hi + 8);  // K = 16hi+8..

      // B fragment (input patch): lane col N = nlo, K = q + 16*hi.
      const int iyd = oy * stride - pad + ty;
      const int ixd = ox * stride - pad + tx;
      int iy = iyd, ix = ixd;
      bool ok = true;
      if (DIL == 2) { ok = (((iyd | ixd) & 1) == 0); iy = iyd >> 1; ix = ixd >> 1; }
      ok = ok && (iy >= 0) && (iy < Hin) && (ix >= 0) && (ix < Win);
      v16h bf;
      if (ok) {
        const _Float16* xr =
            X + (((long)b * Hin + iy) * Win + ix) * Ipad + ic + 16 * hi;
        ((uint4*)&bf)[0] = *(const uint4*)(xr);
        ((uint4*)&bf)[1] = *(const uint4*)(xr + 8);
        if (style16) bf *= sf;                   // modulation folded into B
      } else {
#pragma unroll
        for (int q = 0; q < 16; ++q) bf[q] = (_Float16)0.f;
      }
      acc = __builtin_amdgcn_wmma_f32_16x16x32_f16(false, a, false, bf,
                                                   (short)0, acc, false, false);
    }
  }

  if (n >= HW) return;
  float nz = 0.f;
  if (noise) nz = noise[oy * Wout + ox] * nstr[0];
  __align__(16) _Float16 outh[8];
#pragma unroll
  for (int r = 0; r < 8; ++r) {
    const int oc = ot * 16 + 8 * hi + r;         // D: M = r + 8*hi
    float v = acc[r];
    if (demod) v *= demod[(long)b * Opad + oc];
    v += nz;
    if (bias && oc < Oreal) v += bias[oc];
    if (act) v = lrelu_s2(v);
    outh[r] = (_Float16)v;
    if (Yf && oc < Oreal)
      Yf[(((long)b * Oreal + oc) * Hout + oy) * Wout + ox] = v;
  }
  if (Yh) {
    _Float16* yp =
        Yh + (((long)b * Hout + oy) * Wout + ox) * Opad + ot * 16 + 8 * hi;
    *(uint4*)yp = *(const uint4*)outh;           // 8 contiguous channels
  }
}

// ---- adaptive avg pool (HxH -> 1) over real channels ----------------------
__global__ void k_meanpool(const _Float16* __restrict__ X, float* __restrict__ out,
                           int B, int Cpad, int Creal, int H) {
  int i = blockIdx.x * blockDim.x + threadIdx.x;
  if (i >= B * Creal) return;
  int c = i % Creal, b = i / Creal;
  float s = 0.f;
  for (int y = 0; y < H; ++y)
    for (int x = 0; x < H; ++x)
      s += (float)X[(((long)b * H + y) * H + x) * Cpad + c];
  out[i] = s / (float)(H * H);
}

// ---- wlat = concat(broadcast senc, dlatents) ------------------------------
__global__ void k_wlat(const float* __restrict__ senc, const float* __restrict__ dlat,
                       float* __restrict__ wlat, int B, int NW, int SD, int DD) {
  int D = SD + DD;
  long i = (long)blockIdx.x * blockDim.x + threadIdx.x;
  long total = (long)B * NW * D;
  if (i >= total) return;
  int d = (int)(i % D); long p = i / D;
  int w = (int)(p % NW); int b = (int)(p / NW);
  wlat[i] = (d < SD) ? senc[b * SD + d]
                     : dlat[((long)b * NW + w) * DD + (d - SD)];
}

// ---- style: flrelu(wlat_row @ (mw.T*g), mb); writes f32 + f16, 0 on pads --
__global__ void k_style(const float* __restrict__ wlat, int widx, int NW, int D,
                        const float* __restrict__ mw, const float* __restrict__ mb,
                        float* __restrict__ sty, _Float16* __restrict__ sty16,
                        int B, int Ireal, int Ipad, float g) {
  int i = blockIdx.x * blockDim.x + threadIdx.x;
  if (i >= B * Ipad) return;
  int c = i % Ipad, b = i / Ipad;
  float v = 0.f;
  if (c < Ireal) {
    const float* wr = wlat + ((long)b * NW + widx) * D;
    const float* mr = mw + (long)c * D;
    float s = 0.f;
    for (int d = 0; d < D; ++d) s += wr[d] * mr[d];
    v = lrelu_s2(s * g + mb[c]);
  }
  sty[i] = v;
  sty16[i] = (_Float16)v;
}

// ---- demod[b,o] = rsqrt(sum_i sum_t (w*scale)^2 * style^2 + 1e-8) ---------
__global__ void k_demod(const float* __restrict__ w, const float* __restrict__ sty,
                        float* __restrict__ dem, int B, int O, int Opad,
                        int I, int Ipad, int kk, float scale) {
  int i = blockIdx.x * blockDim.x + threadIdx.x;
  if (i >= B * Opad) return;
  int oc = i % Opad, b = i / Opad;
  float d = 1.f;
  if (oc < O) {
    float acc = 0.f;
    const float* wr = w + (long)oc * I * kk;
    for (int ii = 0; ii < I; ++ii) {
      float w2 = 0.f;
      for (int t = 0; t < kk; ++t) { float ww = wr[ii * kk + t]; w2 += ww * ww; }
      float s = sty[(long)b * Ipad + ii];
      acc += w2 * s * s;
    }
    d = rsqrtf(scale * scale * acc + 1e-8f);
  }
  dem[i] = d;
}

// ---- weight repack: f32 [O,I,kh,kw]*scale -> f16 [kk][Opad][Ipad] ---------
__global__ void k_repack(const float* __restrict__ w, _Float16* __restrict__ wt,
                         int O, int I, int kk, int Opad, int Ipad, float scale) {
  long i = (long)blockIdx.x * blockDim.x + threadIdx.x;
  long total = (long)kk * Opad * Ipad;
  if (i >= total) return;
  int ii = (int)(i % Ipad); long p = i / Ipad;
  int o = (int)(p % Opad); int t = (int)(p / Opad);
  float v = 0.f;
  if (o < O && ii < I) v = w[((long)o * I + ii) * kk + t] * scale;
  wt[i] = (_Float16)v;
}

// ---- skip = torgb + upsample2d(skip): 2x zero-insert, 4x4 FIR * 4 ---------
__global__ void k_skip_up_add(const float* __restrict__ T, const float* __restrict__ S,
                              float* __restrict__ O_, int B, int C, int Hin, int Hout) {
  long i = (long)blockIdx.x * blockDim.x + threadIdx.x;
  long total = (long)B * C * Hout * Hout;
  if (i >= total) return;
  int x = (int)(i % Hout); long p = i / Hout;
  int y = (int)(p % Hout); p /= Hout;
  int c = (int)(p % C); int b = (int)(p / C);
  float acc = 0.f;
#pragma unroll
  for (int ty = 0; ty < 4; ++ty) {
    int uy = y - 2 + ty;
    if (uy < 0 || (uy & 1)) continue;
    int iy = uy >> 1;
    if (iy >= Hin) continue;
#pragma unroll
    for (int tx = 0; tx < 4; ++tx) {
      int ux = x - 2 + tx;
      if (ux < 0 || (ux & 1)) continue;
      int ix = ux >> 1;
      if (ix >= Hin) continue;
      acc += ftap(ty) * ftap(tx) * S[(((long)b * C + c) * Hin + iy) * Hin + ix];
    }
  }
  O_[i] = T[i] + 4.f * acc;
}

// ---------------------------------------------------------------------------
extern "C" void kernel_launch(void* const* d_in, const int* in_sizes, int n_in,
                              void* d_out, int out_size, void* d_ws, size_t ws_size,
                              hipStream_t stream) {
  (void)in_sizes; (void)n_in; (void)out_size; (void)ws_size;
  const int B = 4;
  const float* dlat       = (const float*)d_in[0];
  const float* style_in   = (const float*)d_in[1];
  const float* content_in = (const float*)d_in[2];
  const float* p_input    = (const float*)d_in[3];
  auto F = [&](int i) { return (const float*)d_in[i]; };
  // Input flattening (setup_inputs insertion order, depth-first):
  //  style_enc w/b: 4+2l / 5+2l     content_enc: 16+2l / 17+2l
  //  bottom: 28 w, 29 mw, 30 mb, 31 bias, 32 nstrength, 33 nconst
  //  convs j: 34+6j (w, mw, mb, bias, nstrength, nconst)
  //  trgbs t: 106+4t (w, mw, mb, bias)

  // ---- workspace carve ----
  char* wsb = (char*)d_ws; size_t off = 0;
  auto alloc = [&](size_t bytes) -> void* {
    size_t o = (off + 255) & ~(size_t)255; off = o + bytes; return (void*)(wsb + o);
  };
  const size_t ACT_E = (size_t)B * 257 * 257 * 128;           // largest NHWC f16
  _Float16* ACT0 = (_Float16*)alloc(ACT_E * 2);
  _Float16* ACT1 = (_Float16*)alloc(ACT_E * 2);
  _Float16* ACT2 = (_Float16*)alloc(ACT_E * 2);
  _Float16* WBUF = (_Float16*)alloc((size_t)9 * 512 * 512 * 2);
  static const int contC_[6] = {128, 256, 256, 256, 256, 256};
  static const int contH_[6] = {128, 64, 32, 16, 8, 4};
  _Float16* CONT[6];
  for (int l = 0; l < 6; ++l)
    CONT[l] = (_Float16*)alloc((size_t)B * contH_[l] * contH_[l] * contC_[l] * 2);
  float*     SENC = (float*)alloc((size_t)B * 112 * 4);
  float*     WLAT = (float*)alloc((size_t)B * 14 * 624 * 4);
  float*     STY  = (float*)alloc((size_t)B * 512 * 4);
  _Float16*  STYH = (_Float16*)alloc((size_t)B * 512 * 2);
  float*     DEM  = (float*)alloc((size_t)B * 512 * 4);
  float*     SK0  = (float*)alloc((size_t)B * 3 * 256 * 256 * 4);
  float*     SK1  = (float*)alloc((size_t)B * 3 * 256 * 256 * 4);
  float*     TRGB = (float*)alloc((size_t)B * 3 * 256 * 256 * 4);

  auto conv = [&](const _Float16* Xp, _Float16* Yh, float* Yf,
                  const _Float16* sty16, const float* dem, const float* bias,
                  const float* noise, const float* nstr,
                  int Hin, int Ipad, int Hout, int Opad, int Oreal,
                  int ks, int stride, int pad, int dil, int act) {
    int ntiles = idiv((long)Hout * Hout, 16);
    dim3 g(idiv(ntiles, 8), Opad / 16, B);
    if (ks == 3 && dil == 1)
      k_conv_wmma<3, 1><<<g, THREADS, 0, stream>>>(Xp, WBUF, Yh, Yf, sty16, dem,
          bias, noise, nstr, B, Hin, Hin, Ipad, Hout, Hout, Opad, Oreal,
          stride, pad, act);
    else if (ks == 3 && dil == 2)
      k_conv_wmma<3, 2><<<g, THREADS, 0, stream>>>(Xp, WBUF, Yh, Yf, sty16, dem,
          bias, noise, nstr, B, Hin, Hin, Ipad, Hout, Hout, Opad, Oreal,
          stride, pad, act);
    else
      k_conv_wmma<1, 1><<<g, THREADS, 0, stream>>>(Xp, WBUF, Yh, Yf, sty16, dem,
          bias, noise, nstr, B, Hin, Hin, Ipad, Hout, Hout, Opad, Oreal,
          stride, pad, act);
  };
  auto fir = [&](const _Float16* Xp, _Float16* Y, int Hin, int Cpad, int Hout, int pad,
                 float gain, const float* noise, const float* nstr,
                 const float* bias, int Creal, int act) {
    long tot = (long)B * Hout * Hout * Cpad;
    k_fir_h<<<idiv(tot, THREADS), THREADS, 0, stream>>>(Xp, Y, B, Hin, Hin, Cpad,
        Hout, Hout, pad, gain, noise, nstr, bias, Creal, act);
  };
  auto repack = [&](const float* w, int O, int I, int kk, int Opad, int Ipad, float sc) {
    long tot = (long)kk * Opad * Ipad;
    k_repack<<<idiv(tot, THREADS), THREADS, 0, stream>>>(w, WBUF, O, I, kk, Opad, Ipad, sc);
  };
  auto modprep = [&](int widx, const float* mw, const float* mb, int Ireal, int Ipad,
                     const float* w, int O, int Opad, int kk, float sc, bool dem) {
    k_style<<<idiv((long)B * Ipad, THREADS), THREADS, 0, stream>>>(
        WLAT, widx, 14, 624, mw, mb, STY, STYH, B, Ireal, Ipad, 1.f / sqrtf(624.f));
    if (dem)
      k_demod<<<idiv((long)B * Opad, THREADS), THREADS, 0, stream>>>(
          w, STY, DEM, B, O, Opad, Ireal, Ipad, kk, sc);
    repack(w, O, Ireal, kk, Opad, Ipad, sc);
  };

  // ================= style encoder -> senc [B,112] =========================
  {
    long tot = (long)B * 256 * 256 * 32;
    k_pack<<<idiv(tot, THREADS), THREADS, 0, stream>>>(style_in, ACT0, B, 3, 256, 256,
                                                       32, 0, 32, 0);
    static const int seI[6] = {3, 112, 112, 112, 112, 112};
    int H = 256, Cp = 32;
    _Float16* sp = ACT0; _Float16* sq = ACT2;
    for (int l = 0; l < 6; ++l) {
      repack(F(4 + 2 * l), 112, seI[l], 9, 128, Cp, 1.f / sqrtf((float)seI[l] * 9.f));
      fir(sp, ACT1, H, Cp, H + 1, 2, 1.f, nullptr, nullptr, nullptr, 0, 0);
      conv(ACT1, sq, nullptr, nullptr, nullptr, F(5 + 2 * l), nullptr, nullptr,
           H + 1, Cp, H / 2, 128, 112, 3, 2, 0, 1, 1);
      _Float16* t = sp; sp = sq; sq = t;
      H /= 2; Cp = 128;
    }
    k_meanpool<<<idiv((long)B * 112, THREADS), THREADS, 0, stream>>>(sp, SENC, B, 128, 112, 4);
  }

  // ================= content encoder (keep all scales) =====================
  {
    long tot = (long)B * 256 * 256 * 32;
    k_pack<<<idiv(tot, THREADS), THREADS, 0, stream>>>(content_in, ACT0, B, 4, 256, 256,
                                                       32, 0, 32, 0);
    static const int ceI[6]  = {4, 128, 256, 256, 256, 256};
    static const int ceIp[6] = {32, 128, 256, 256, 256, 256};
    static const int ceO[6]  = {128, 256, 256, 256, 256, 256};
    int H = 256; const _Float16* src = ACT0;
    for (int l = 0; l < 6; ++l) {
      repack(F(16 + 2 * l), ceO[l], ceI[l], 9, ceO[l], ceIp[l],
             1.f / sqrtf((float)ceI[l] * 9.f));
      fir(src, ACT1, H, ceIp[l], H + 1, 2, 1.f, nullptr, nullptr, nullptr, 0, 0);
      conv(ACT1, CONT[l], nullptr, nullptr, nullptr, F(17 + 2 * l), nullptr, nullptr,
           H + 1, ceIp[l], H / 2, ceO[l], ceO[l], 3, 2, 0, 1, 1);
      src = CONT[l]; H /= 2;
    }
  }

  // ================= wlat ==================================================
  k_wlat<<<idiv((long)B * 14 * 624, THREADS), THREADS, 0, stream>>>(SENC, dlat, WLAT,
                                                                    B, 14, 112, 512);

  // ================= synthesis =============================================
  // bottom: x = concat(const input, content 4x4) -> modulated 3x3 conv
  {
    long t1 = (long)B * 4 * 4 * 256;
    k_pack<<<idiv(t1, THREADS), THREADS, 0, stream>>>(p_input, ACT0, B, 256, 4, 4,
                                                      512, 0, 256, 1);
    k_copy_h<<<idiv(t1, THREADS), THREADS, 0, stream>>>(CONT[5], ACT0, B, 4, 4,
                                                        256, 512, 256, 256);
    modprep(0, F(29), F(30), 512, 512, F(28), 256, 256, 9,
            1.f / sqrtf(512.f * 9.f), true);
    conv(ACT0, ACT1, nullptr, STYH, DEM, F(31), F(33), F(32),
         4, 512, 4, 256, 256, 3, 1, 1, 1, 1);
  }
  _Float16* X  = ACT1; int Xc = 256; int H = 4;
  _Float16* FA = ACT0; _Float16* FB = ACT2;

  // trgb0 -> skip (no upsample-add for the first one)
  modprep(1, F(107), F(108), 256, 256, F(106), 3, 16, 1, 1.f / sqrtf(256.f), false);
  conv(X, nullptr, SK0, STYH, nullptr, F(109), nullptr, nullptr,
       4, 256, 4, 16, 3, 1, 1, 0, 1, 0);

  float* skip = SK0; float* skipN = SK1;
  static const int inC_[12]  = {256, 512, 512, 512, 512, 512, 512, 512, 512, 256, 256, 128};
  static const int outC_[12] = {256, 512, 256, 512, 256, 512, 256, 512, 128, 256, 128, 128};
  static const int cIdx[5] = {4, 3, 2, 1, 0};

  for (int ri = 0; ri < 6; ++ri) {
    const int j0 = 2 * ri, j1 = 2 * ri + 1;
    const int b0 = 34 + 6 * j0, b1 = 34 + 6 * j1;
    const int H2 = 2 * H;

    // conv0 (up): modulate, 3x3 on 2x zero-dilated input (pad 2), demod epi
    modprep(j0 + 1, F(b0 + 1), F(b0 + 2), inC_[j0], Xc, F(b0 + 0),
            outC_[j0], outC_[j0], 9, 1.f / sqrtf((float)inC_[j0] * 9.f), true);
    conv(X, FA, nullptr, STYH, DEM, nullptr, nullptr, nullptr,
         H, Xc, H2 + 1, outC_[j0], outC_[j0], 3, 1, 2, 2, 0);
    // FIR(pad 1, gain 4) + fused noise/bias/lrelu
    fir(FA, FB, H2 + 1, outC_[j0], H2, 1, 4.f, F(b0 + 5), F(b0 + 4), F(b0 + 3),
        outC_[j0], 1);

    _Float16* oldX = X;
    _Float16* x2; int x2c;
    if (ri < 5) {  // concat with content encoder features
      int cc = contC_[cIdx[ri]];
      x2c = outC_[j0] + cc;
      long tc = (long)B * H2 * H2 * outC_[j0];
      k_copy_h<<<idiv(tc, THREADS), THREADS, 0, stream>>>(FB, oldX, B, H2, H2,
          outC_[j0], x2c, 0, outC_[j0]);
      long tc2 = (long)B * H2 * H2 * cc;
      k_copy_h<<<idiv(tc2, THREADS), THREADS, 0, stream>>>(CONT[cIdx[ri]], oldX, B, H2, H2,
          cc, x2c, outC_[j0], cc);
      x2 = oldX;
    } else { x2 = FB; x2c = outC_[j0]; }

    // conv1: modulated 3x3 pad 1, demod + noise + bias + lrelu epilogue
    modprep(j1 + 1, F(b1 + 1), F(b1 + 2), inC_[j1], x2c, F(b1 + 0),
            outC_[j1], outC_[j1], 9, 1.f / sqrtf((float)inC_[j1] * 9.f), true);
    conv(x2, FA, nullptr, STYH, DEM, F(b1 + 3), F(b1 + 5), F(b1 + 4),
         H2, x2c, H2, outC_[j1], outC_[j1], 3, 1, 1, 1, 1);
    X = FA; FA = oldX;  // rotate: free = {oldX, FB}

    // toRGB (1x1 modulated, no demod, bias, no act) -> fp32 NCHW tmp
    const int tb = 106 + 4 * (ri + 1);
    modprep(2 * ri + 3, F(tb + 1), F(tb + 2), outC_[j1], outC_[j1], F(tb + 0),
            3, 16, 1, 1.f / sqrtf((float)outC_[j1]), false);
    conv(X, nullptr, TRGB, STYH, nullptr, F(tb + 3), nullptr, nullptr,
         H2, outC_[j1], H2, 16, 3, 1, 1, 0, 1, 0);

    // skip = torgb + upsample2d(skip)
    float* sout = (ri == 5) ? (float*)d_out : skipN;
    long ts = (long)B * 3 * H2 * H2;
    k_skip_up_add<<<idiv(ts, THREADS), THREADS, 0, stream>>>(TRGB, skip, sout,
                                                             B, 3, H, H2);
    if (ri != 5) { float* tt = skip; skip = skipN; skipN = tt; }
    H = H2; Xc = outC_[j1];
  }
}
